// NAOMIModel_12592844112395
// MI455X (gfx1250) — matrix operand order repaired and checked by reading, not compiled, not run on hardware
//
#include <hip/hip_runtime.h>
#include <hip/hip_bf16.h>
#include <math.h>

// ---------------------------------------------------------------------------
// NAOMI model for MI455X (gfx1250, wave32, WMMA).
// Fwd/Bwd stacked LSTM: one 1024-thread workgroup per direction, fp16
// activations double-buffered in LDS, fp16 weights streamed from L2 with
// depth-2 fragment prefetch and 2-way M-tile B-reuse into
// v_wmma_f32_16x16x32_f16 gate GEMMs. c-state lives in a coalesced f32
// workspace to keep VGPR pressure under the 1024-thread cap.
// Decoder heads: one block per (head,timestep); activations staged into LDS
// via global_load_async_to_lds_b128 (ASYNCcnt), WMMA GEMMs + fused SSE loss.
// Weight accesses use an asm-laundered zero OFFSET (not pointer) so loads
// stay global_load_b128 (LOADcnt-only) while LICM cannot hoist/spill them.
// ---------------------------------------------------------------------------

typedef __attribute__((ext_vector_type(16))) _Float16 v16h;
typedef __attribute__((ext_vector_type(8)))  _Float16 v8h;
typedef __attribute__((ext_vector_type(8)))  float    v8f;

#define T_SEQ 512
#define BATCH 64
#define DDIM  64
#define HDIM  512

// Layer-0 K width (both dirs): [x | zero-pad to 96 | h(512)] = 608
#define K0   608
#define KT0  19          // 608/32 k-tiles
#define K1   1024        // layer1: [h0' | h1]
#define KT1  32

#define XW0  1120        // xh0 row width: 96 + 2*512 (double-buffered h0)
#define XW1  1024        // xh1 row width: 2*512 (double-buffered h1)

// ---- workspace layout (bytes, all 256-aligned) ----
#define HF_BYTES   ((size_t)T_SEQ * BATCH * HDIM * 2)          // 33,554,432
#define HB_BYTES   ((size_t)(T_SEQ - 1) * BATCH * HDIM * 2)    // 33,488,896
#define W0_BYTES   ((size_t)4 * HDIM * K0 * 2)                 //  2,490,368
#define W1_BYTES   ((size_t)4 * HDIM * K1 * 2)                 //  4,194,304
#define DW_BYTES   ((size_t)5 * HDIM * 1024 * 2)               //  5,242,880
#define MW_BYTES   ((size_t)5 * DDIM * HDIM * 2)               //    327,680
#define C_BYTES    ((size_t)2 * 2 * BATCH * HDIM * 4)          //    524,288

#define HF_OFF   ((size_t)0)
#define HB_OFF   (HF_OFF + HF_BYTES)
#define FW0_OFF  (HB_OFF + HB_BYTES)
#define BW0_OFF  (FW0_OFF + W0_BYTES)
#define FW1_OFF  (BW0_OFF + W0_BYTES)
#define BW1_OFF  (FW1_OFF + W1_BYTES)
#define DW_OFF   (BW1_OFF + W1_BYTES)
#define MW_OFF   (DW_OFF + DW_BYTES)
#define C_OFF    (MW_OFF + MW_BYTES)

// Opaque zero: defeats LICM/CSE of loop-invariant weight fragment loads while
// preserving the base pointer's inferred global address space.
__device__ __forceinline__ size_t opaque_zero() {
  size_t z = 0;
  asm volatile("" : "+s"(z));
  return z;
}

// ---------------------------------------------------------------------------
// Fragment loaders (ISA 7.12.2 layouts, wave32)
// ---------------------------------------------------------------------------

// A: 16x32 fp16 from a 32-half window `p` (row start + k column, 4B aligned).
// VGPR g0..7 hold K=2g+16*sel (contiguous, matches load_bT).
__device__ __forceinline__ v16h load_a_win(const _Float16* __restrict__ p, int sel) {
  const uint32_t* u = (const uint32_t*)p;
  const int s4 = sel * 4;
  union { uint32_t u[8]; v16h v; } cv;
#pragma unroll
  for (int g = 0; g < 4; ++g) cv.u[g] = u[g + 2 * s4];
#pragma unroll
  for (int g = 0; g < 4; ++g) cv.u[4 + g] = u[4 + g + 2 * s4];
  return cv.v;
}

// B: 32x16 fp16 from a transposed (N,K) row-major weight; lane holds column
// n = lane&15, K = 16*sel .. +15 contiguous (two 16-byte loads).
__device__ __forceinline__ v16h load_bT(const _Float16* __restrict__ wT,
                                        int ldk, int nrow_base, int ktile, int lane) {
  const int n  = nrow_base + (lane & 15);
  const int k0 = ktile * 32 + ((lane >> 4) << 4);
  const _Float16* p = wT + (size_t)n * ldk + k0;
  union { v8h h[2]; v16h v; } cv;
  cv.h[0] = *(const v8h*)(p);
  cv.h[1] = *(const v8h*)(p + 8);
  return cv.v;
}

__device__ __forceinline__ float sigf(float x) { return 1.0f / (1.0f + __expf(-x)); }

// ---------------------------------------------------------------------------
// Weight conversion: fp16 [Wih | 0-pad | Whh] concat matrices (K0=608 wide for
// layer 0, K1=1024 for layer 1) and transposed decoder/mean weights.
// ---------------------------------------------------------------------------
__global__ void convert_kernel(
    const float* __restrict__ fWih0, const float* __restrict__ fWhh0,
    const float* __restrict__ fWih1, const float* __restrict__ fWhh1,
    const float* __restrict__ bWih0, const float* __restrict__ bWhh0,
    const float* __restrict__ bWih1, const float* __restrict__ bWhh1,
    const float* __restrict__ decW,  const float* __restrict__ meanW,
    _Float16* __restrict__ fW0, _Float16* __restrict__ bW0,
    _Float16* __restrict__ fW1, _Float16* __restrict__ bW1,
    _Float16* __restrict__ dWT, _Float16* __restrict__ mWT) {
  const long N0 = 2048L * K0, N1 = 2048L * K1;
  const long Nd = 5L * HDIM * 1024, Nm = 5L * DDIM * HDIM;
  const long total = 2 * N0 + 2 * N1 + Nd + Nm;
  const long stride = (long)gridDim.x * blockDim.x;
  for (long idx = (long)blockIdx.x * blockDim.x + threadIdx.x; idx < total; idx += stride) {
    long x = idx;
    if (x < N0) {  // fwd layer0: x cols 0..63, pad 64..95, Whh 96..607
      int r = (int)(x / K0), c = (int)(x % K0);
      float v = (c < 64) ? fWih0[(long)r * 64 + c]
                         : ((c < 96) ? 0.0f : fWhh0[(long)r * 512 + (c - 96)]);
      fW0[x] = (_Float16)v;
      continue;
    }
    x -= N0;
    if (x < N0) {  // bwd layer0: x cols 0..64, pad 65..95, Whh 96..607
      int r = (int)(x / K0), c = (int)(x % K0);
      float v = (c < 65) ? bWih0[(long)r * 65 + c]
                         : ((c < 96) ? 0.0f : bWhh0[(long)r * 512 + (c - 96)]);
      bW0[x] = (_Float16)v;
      continue;
    }
    x -= N0;
    if (x < N1) {
      int r = (int)(x / K1), c = (int)(x % K1);
      fW1[x] = (_Float16)((c < 512) ? fWih1[(long)r * 512 + c] : fWhh1[(long)r * 512 + (c - 512)]);
      continue;
    }
    x -= N1;
    if (x < N1) {
      int r = (int)(x / K1), c = (int)(x % K1);
      bW1[x] = (_Float16)((c < 512) ? bWih1[(long)r * 512 + c] : bWhh1[(long)r * 512 + (c - 512)]);
      continue;
    }
    x -= N1;
    if (x < Nd) {  // dWT[l][h=512][i=1024] = decW[l][i][h]
      int l = (int)(x / (512L * 1024));
      long rem = x % (512L * 1024);
      int h = (int)(rem / 1024), i2 = (int)(rem % 1024);
      dWT[x] = (_Float16)decW[(long)l * 1024 * 512 + (long)i2 * 512 + h];
      continue;
    }
    x -= Nd;
    {  // mWT[l][d=64][h=512] = meanW[l][h][d]
      int l = (int)(x / (64L * 512));
      long rem = x % (64L * 512);
      int d = (int)(rem / 512), h = (int)(rem % 512);
      mWT[x] = (_Float16)meanW[(long)l * 512 * 64 + (long)h * 64 + d];
    }
  }
}

__global__ void init_kernel(float* __restrict__ loss, _Float16* __restrict__ hb) {
  long idx = (long)blockIdx.x * blockDim.x + threadIdx.x;
  if (idx == 0) *loss = 0.0f;
  // hb_by_key[T-2] (= hb[510], the last slot) is the zero initial h1 of the backward LSTM.
  const long base = (long)(T_SEQ - 2) * BATCH * HDIM;
  const long stride = (long)gridDim.x * blockDim.x;
  for (long i = idx; i < (long)BATCH * HDIM; i += stride) hb[base + i] = (_Float16)0.0f;
}

// ---------------------------------------------------------------------------
// Stacked 2-layer LSTM recurrence. grid.x = 2 (0 = fwd, 1 = bwd),
// block = 1024 threads (32 waves). Wave w owns N-tile w of H for both layers.
// M-tiles processed in pairs (2 accumulator chains share each B fragment),
// gates fully unrolled, depth-2 B prefetch. c-state: coalesced f32 workspace.
// Dynamic LDS: xh0 (64 x 1120 fp16) + xh1 (64 x 1024 fp16) = 274,432 B.
// ---------------------------------------------------------------------------
__global__ void __launch_bounds__(1024) lstm_kernel(
    const float* __restrict__ gt, const float* __restrict__ data,
    const _Float16* __restrict__ fW0, const float* __restrict__ fb0,
    const _Float16* __restrict__ fW1, const float* __restrict__ fb1,
    const _Float16* __restrict__ bW0, const float* __restrict__ bb0,
    const _Float16* __restrict__ bW1, const float* __restrict__ bb1,
    float* __restrict__ cws,
    _Float16* __restrict__ hf, _Float16* __restrict__ hb) {
  extern __shared__ char smem[];
  _Float16* xh0 = (_Float16*)smem;                           // 64 x 1120
  _Float16* xh1 = (_Float16*)(smem + (size_t)64 * XW0 * 2);  // 64 x 1024

  const int dir  = blockIdx.x;
  const int tid  = threadIdx.x;
  const int wave = tid >> 5;
  const int lane = tid & 31;
  const int laneN = lane & 15;
  const int sel   = lane >> 4;
  const int ntile = wave;  // 0..31 -> N-tile of H

  const _Float16* W0 = dir ? bW0 : fW0;
  const _Float16* W1 = dir ? bW1 : fW1;
  const float* b0 = dir ? bb0 : fb0;
  const float* b1 = dir ? bb1 : fb1;
  const int Din   = dir ? 65 : 64;
  const int steps = dir ? (T_SEQ - 1) : T_SEQ;

  float* cL0 = cws + (size_t)dir * 65536;          // [q][r][tid] f32
  float* cL1 = cL0 + 32768;

  for (int i = tid; i < 64 * XW0; i += 1024) xh0[i] = (_Float16)0.0f;
  for (int i = tid; i < 64 * XW1; i += 1024) xh1[i] = (_Float16)0.0f;
  __syncthreads();

  for (int t = 0; t < steps; ++t) {
    const int buf = t & 1, nxt = buf ^ 1;
    const int h0cur = 96 + buf * 512, h0nxt = 96 + nxt * 512;
    const int h1cur = buf * 512,      h1nxt = nxt * 512;

    // ---- stage x_t into xh0[:, 0:Din] (f32 -> fp16) ----
    {
      const float* src = dir ? (data + (size_t)(T_SEQ - 1 - t) * BATCH * 65)
                             : (gt + (size_t)t * BATCH * DDIM);
      const int total = BATCH * Din;
      for (int i = tid; i < total; i += 1024) {
        int b = i / Din, d = i - b * Din;
        xh0[b * XW0 + d] = (_Float16)src[i];
      }
    }
    __syncthreads();

    // ---- layer 0: gates = [x | h0@buf] @ W0^T + b0 ; h0' -> h0@nxt ----
#pragma unroll 1
    for (int qp = 0; qp < 2; ++qp) {
      const _Float16* W = W0 + opaque_zero();  // keep B loads in this pass
      const _Float16* ar0 = xh0 + (size_t)((qp * 2 + 0) * 16 + laneN) * XW0;
      const _Float16* ar1 = xh0 + (size_t)((qp * 2 + 1) * 16 + laneN) * XW0;
      v8f acc[4][2];
#pragma unroll
      for (int g = 0; g < 4; ++g) {
        float bv = b0[g * HDIM + ntile * 16 + laneN];
#pragma unroll
        for (int r = 0; r < 8; ++r) { acc[g][0][r] = bv; acc[g][1][r] = bv; }
      }
#pragma unroll
      for (int g = 0; g < 4; ++g) {
        const int nb = g * HDIM + ntile * 16;
        v16h bA = load_bT(W, K0, nb, 0, lane);
        v16h bB = load_bT(W, K0, nb, 1, lane);
        for (int k = 0; k < KT0; ++k) {
          const int kp = (k + 2 < KT0) ? (k + 2) : k;
          v16h bC = load_bT(W, K0, nb, kp, lane);
          const int pcol = (k < 3) ? (k * 32) : (h0cur + (k - 3) * 32);
          v16h a0 = load_a_win(ar0 + pcol, sel);
          v16h a1 = load_a_win(ar1 + pcol, sel);
          acc[g][0] = __builtin_amdgcn_wmma_f32_16x16x32_f16(
              false, a0, false, bA, (short)0, acc[g][0], false, false);
          acc[g][1] = __builtin_amdgcn_wmma_f32_16x16x32_f16(
              false, a1, false, bA, (short)0, acc[g][1], false, false);
          bA = bB; bB = bC;
        }
      }
#pragma unroll
      for (int qq = 0; qq < 2; ++qq) {
        const int q = qp * 2 + qq;
        float* cb = cL0 + (size_t)q * 8 * 1024 + tid;
#pragma unroll
        for (int r = 0; r < 8; ++r) {
          float cprev = t ? cb[r * 1024] : 0.0f;
          float iv = sigf(acc[0][qq][r]);
          float fv = sigf(acc[1][qq][r]);
          float gv = tanhf(acc[2][qq][r]);
          float ov = sigf(acc[3][qq][r]);
          float c = fv * cprev + iv * gv;
          cb[r * 1024] = c;
          int M = q * 16 + r + sel * 8;
          int N = ntile * 16 + laneN;
          xh0[M * XW0 + h0nxt + N] = (_Float16)(ov * tanhf(c));
        }
      }
    }
    __syncthreads();

    // ---- layer 1: gates = [h0@nxt | h1@buf] @ W1^T + b1 ; h1' -> h1@nxt ----
#pragma unroll 1
    for (int qp = 0; qp < 2; ++qp) {
      const _Float16* W = W1 + opaque_zero();
      const _Float16* ar0a = xh0 + (size_t)((qp * 2 + 0) * 16 + laneN) * XW0 + h0nxt;
      const _Float16* ar0b = xh1 + (size_t)((qp * 2 + 0) * 16 + laneN) * XW1 + h1cur;
      const _Float16* ar1a = xh0 + (size_t)((qp * 2 + 1) * 16 + laneN) * XW0 + h0nxt;
      const _Float16* ar1b = xh1 + (size_t)((qp * 2 + 1) * 16 + laneN) * XW1 + h1cur;
      v8f acc[4][2];
#pragma unroll
      for (int g = 0; g < 4; ++g) {
        float bv = b1[g * HDIM + ntile * 16 + laneN];
#pragma unroll
        for (int r = 0; r < 8; ++r) { acc[g][0][r] = bv; acc[g][1][r] = bv; }
      }
#pragma unroll
      for (int g = 0; g < 4; ++g) {
        const int nb = g * HDIM + ntile * 16;
        v16h bA = load_bT(W, K1, nb, 0, lane);
        v16h bB = load_bT(W, K1, nb, 1, lane);
        for (int k = 0; k < KT1; ++k) {
          const int kp = (k + 2 < KT1) ? (k + 2) : k;
          v16h bC = load_bT(W, K1, nb, kp, lane);
          v16h a0 = load_a_win((k < 16) ? (ar0a + k * 32) : (ar0b + (k - 16) * 32), sel);
          v16h a1 = load_a_win((k < 16) ? (ar1a + k * 32) : (ar1b + (k - 16) * 32), sel);
          acc[g][0] = __builtin_amdgcn_wmma_f32_16x16x32_f16(
              false, a0, false, bA, (short)0, acc[g][0], false, false);
          acc[g][1] = __builtin_amdgcn_wmma_f32_16x16x32_f16(
              false, a1, false, bA, (short)0, acc[g][1], false, false);
          bA = bB; bB = bC;
        }
      }
#pragma unroll
      for (int qq = 0; qq < 2; ++qq) {
        const int q = qp * 2 + qq;
        float* cb = cL1 + (size_t)q * 8 * 1024 + tid;
#pragma unroll
        for (int r = 0; r < 8; ++r) {
          float cprev = t ? cb[r * 1024] : 0.0f;
          float iv = sigf(acc[0][qq][r]);
          float fv = sigf(acc[1][qq][r]);
          float gv = tanhf(acc[2][qq][r]);
          float ov = sigf(acc[3][qq][r]);
          float c = fv * cprev + iv * gv;
          cb[r * 1024] = c;
          int M = q * 16 + r + sel * 8;
          int N = ntile * 16 + laneN;
          _Float16 hv = (_Float16)(ov * tanhf(c));
          xh1[M * XW1 + h1nxt + N] = hv;
          if (!dir) {
            hf[(size_t)t * (BATCH * HDIM) + M * HDIM + N] = hv;               // h_fwd[t]
          } else if (t < T_SEQ - 2) {
            hb[(size_t)(T_SEQ - 3 - t) * (BATCH * HDIM) + M * HDIM + N] = hv; // hb_by_key
          }
        }
      }
    }
    __syncthreads();
  }
}

// ---------------------------------------------------------------------------
// Decoder heads. One block per (head l, timestep i); 256 threads (8 waves).
// x = [hf[i] | hb[i+2s-2]] async-staged into LDS; dec = relu(x @ dW + db)
// staged fp16 in LDS; mean = dec @ mW + mb; fused SSE vs gt[s+i].
// B fragments shared across all M-tile accumulators (4x / 2x reuse).
// Dynamic LDS: xs (64 x 1024 fp16) + decs (64 x 512 fp16) = 196,608 B.
// ---------------------------------------------------------------------------
__global__ void __launch_bounds__(256) decoder_kernel(
    const _Float16* __restrict__ hf, const _Float16* __restrict__ hb,
    const _Float16* __restrict__ dWT, const float* __restrict__ db,
    const _Float16* __restrict__ mWT, const float* __restrict__ mb,
    const float* __restrict__ gt, float* __restrict__ loss) {
  extern __shared__ char smem[];
  _Float16* xs   = (_Float16*)smem;                            // 64 x 1024
  _Float16* decs = (_Float16*)(smem + (size_t)64 * 1024 * 2);  // 64 x 512
  __shared__ float red[256];

  const int ncnt[5] = {511, 509, 505, 497, 481};
  int bid = blockIdx.x;
  int l = 0, off = 0;
  while (l < 4 && bid >= off + ncnt[l]) { off += ncnt[l]; ++l; }
  const int i = bid - off;
  const int s = 1 << l;

  const int tid  = threadIdx.x;
  const int wave = tid >> 5;
  const int lane = tid & 31;
  const int laneN = lane & 15;
  const int sel   = lane >> 4;

  const _Float16* xa = hf + (size_t)i * (BATCH * HDIM);
  const _Float16* xb = hb + (size_t)(i + 2 * s - 2) * (BATCH * HDIM);

  // ---- async-stage x = [xa | xb] into LDS (64 rows x 1024 halves) ----
  for (int c = tid; c < 64 * 128; c += 256) {        // 128 x 16B chunks / row
    const int row = c >> 7, cc = c & 127;
    const _Float16* src = (cc < 64) ? (xa + (size_t)row * HDIM + cc * 8)
                                    : (xb + (size_t)row * HDIM + (cc - 64) * 8);
    uint32_t loff = (uint32_t)(size_t)(const void*)(xs + row * 1024 + cc * 8);
    unsigned long long ga = (unsigned long long)(size_t)src;
    asm volatile("global_load_async_to_lds_b128 %0, %1, off"
                 :: "v"(loff), "v"(ga) : "memory");
  }
  asm volatile("s_wait_asynccnt 0" ::: "memory");
  __syncthreads();

  // ---- stage 1: dec (64 x 512); per N-tile, B shared by 4 M-tile accs ----
  const _Float16* DW = dWT + (size_t)l * HDIM * 1024;
#pragma unroll 1
  for (int nt = 0; nt < 4; ++nt) {
    const int ntile = wave * 4 + nt;
    const int nb = ntile * 16;
    const _Float16* W = DW + opaque_zero();
    const _Float16* ar[4];
#pragma unroll
    for (int m = 0; m < 4; ++m) ar[m] = xs + (size_t)(m * 16 + laneN) * 1024;
    v8f acc[4];
    {
      float bv = db[l * HDIM + nb + laneN];
#pragma unroll
      for (int m = 0; m < 4; ++m)
#pragma unroll
        for (int r = 0; r < 8; ++r) acc[m][r] = bv;
    }
    v16h bA = load_bT(W, 1024, nb, 0, lane);
    v16h bB = load_bT(W, 1024, nb, 1, lane);
    for (int k = 0; k < 32; ++k) {
      const int kp = (k + 2 < 32) ? (k + 2) : k;
      v16h bC = load_bT(W, 1024, nb, kp, lane);
#pragma unroll
      for (int m = 0; m < 4; ++m) {
        v16h a = load_a_win(ar[m] + k * 32, sel);
        acc[m] = __builtin_amdgcn_wmma_f32_16x16x32_f16(
            false, a, false, bA, (short)0, acc[m], false, false);
      }
      bA = bB; bB = bC;
    }
#pragma unroll
    for (int m = 0; m < 4; ++m)
#pragma unroll
      for (int r = 0; r < 8; ++r) {
        float v = fmaxf(acc[m][r], 0.0f);
        int M = m * 16 + r + sel * 8;
        int N = nb + laneN;
        decs[M * HDIM + N] = (_Float16)v;
      }
  }
  __syncthreads();

  // ---- stage 2: mean (64 x 64) + fused SSE; B shared by 2 M-tile accs ----
  const _Float16* MW = mWT + (size_t)l * DDIM * HDIM;
  const float* g = gt + (size_t)(s + i) * (BATCH * DDIM);
  float lsum = 0.0f;
  {
    const int nt2 = wave & 3;       // N-tile 0..3
    const int mh  = wave >> 2;      // M-tile half 0..1
    const int nb = nt2 * 16;
    const _Float16* W = MW + opaque_zero();
    const _Float16* ar0 = decs + (size_t)((mh * 2 + 0) * 16 + laneN) * HDIM;
    const _Float16* ar1 = decs + (size_t)((mh * 2 + 1) * 16 + laneN) * HDIM;
    v8f acc[2];
    {
      float bv = mb[l * DDIM + nb + laneN];
#pragma unroll
      for (int r = 0; r < 8; ++r) { acc[0][r] = bv; acc[1][r] = bv; }
    }
    v16h bA = load_bT(W, HDIM, nb, 0, lane);
    v16h bB = load_bT(W, HDIM, nb, 1, lane);
    for (int k = 0; k < 16; ++k) {
      const int kp = (k + 2 < 16) ? (k + 2) : k;
      v16h bC = load_bT(W, HDIM, nb, kp, lane);
      v16h a0 = load_a_win(ar0 + k * 32, sel);
      v16h a1 = load_a_win(ar1 + k * 32, sel);
      acc[0] = __builtin_amdgcn_wmma_f32_16x16x32_f16(
          false, a0, false, bA, (short)0, acc[0], false, false);
      acc[1] = __builtin_amdgcn_wmma_f32_16x16x32_f16(
          false, a1, false, bA, (short)0, acc[1], false, false);
      bA = bB; bB = bC;
    }
#pragma unroll
    for (int mm = 0; mm < 2; ++mm)
#pragma unroll
      for (int r = 0; r < 8; ++r) {
        int M = (mh * 2 + mm) * 16 + r + sel * 8;
        int N = nb + laneN;
        float d = acc[mm][r] - g[M * DDIM + N];
        lsum += d * d;
      }
  }

  red[tid] = lsum;
  __syncthreads();
  for (int st = 128; st > 0; st >>= 1) {
    if (tid < st) red[tid] += red[tid + st];
    __syncthreads();
  }
  if (tid == 0) atomicAdd(loss, red[0] * (1.0f / (512.0f * 64.0f)));
}

// ---------------------------------------------------------------------------
extern "C" void kernel_launch(void* const* d_in, const int* in_sizes, int n_in,
                              void* d_out, int out_size, void* d_ws, size_t ws_size,
                              hipStream_t stream) {
  (void)in_sizes; (void)n_in; (void)out_size; (void)ws_size;
  const float* data  = (const float*)d_in[0];
  const float* gt    = (const float*)d_in[1];
  const float* fWih0 = (const float*)d_in[2];
  const float* fWhh0 = (const float*)d_in[3];
  const float* fb0   = (const float*)d_in[4];
  const float* fWih1 = (const float*)d_in[5];
  const float* fWhh1 = (const float*)d_in[6];
  const float* fb1   = (const float*)d_in[7];
  const float* bWih0 = (const float*)d_in[8];
  const float* bWhh0 = (const float*)d_in[9];
  const float* bb0   = (const float*)d_in[10];
  const float* bWih1 = (const float*)d_in[11];
  const float* bWhh1 = (const float*)d_in[12];
  const float* bb1   = (const float*)d_in[13];
  const float* decW  = (const float*)d_in[14];
  const float* decb  = (const float*)d_in[15];
  const float* meanW = (const float*)d_in[16];
  const float* meanb = (const float*)d_in[17];

  char* ws = (char*)d_ws;
  _Float16* hf  = (_Float16*)(ws + HF_OFF);
  _Float16* hb  = (_Float16*)(ws + HB_OFF);
  _Float16* fW0 = (_Float16*)(ws + FW0_OFF);
  _Float16* bW0 = (_Float16*)(ws + BW0_OFF);
  _Float16* fW1 = (_Float16*)(ws + FW1_OFF);
  _Float16* bW1 = (_Float16*)(ws + BW1_OFF);
  _Float16* dWT = (_Float16*)(ws + DW_OFF);
  _Float16* mWT = (_Float16*)(ws + MW_OFF);
  float*    cws = (float*)(ws + C_OFF);
  float* loss = (float*)d_out;

  convert_kernel<<<4096, 256, 0, stream>>>(fWih0, fWhh0, fWih1, fWhh1,
                                           bWih0, bWhh0, bWih1, bWhh1,
                                           decW, meanW,
                                           fW0, bW0, fW1, bW1, dWT, mWT);
  init_kernel<<<64, 256, 0, stream>>>(loss, hb);

  const size_t lstm_lds = (size_t)64 * XW0 * 2 + (size_t)64 * XW1 * 2;  // 274,432 B
  lstm_kernel<<<2, 1024, lstm_lds, stream>>>(gt, data,
                                             fW0, fb0, fW1, fb1,
                                             bW0, bb0, bW1, bb1,
                                             cws, hf, hb);

  const size_t dec_lds = (size_t)64 * 1024 * 2 + (size_t)64 * HDIM * 2;  // 196,608 B
  decoder_kernel<<<2503, 256, dec_lds, stream>>>(hf, hb, dWT, decb, mWT, meanb,
                                                 gt, loss);
}
